// DGLHGNNConv_27831388078182
// MI455X (gfx1250) — compile-verified, matrix operand order checked by reading, not run on hardware
//
#include <hip/hip_runtime.h>
#include <hip/hip_bf16.h>

// ---------------------------------------------------------------------------
// Problem constants (from reference)
// ---------------------------------------------------------------------------
#define IN_CH   256
#define OUT_CH  64

typedef __bf16 bf16_t;
typedef __attribute__((ext_vector_type(16))) __bf16 v16bf;
typedef __attribute__((ext_vector_type(8)))  __bf16 v8bf;
typedef __attribute__((ext_vector_type(8)))  float  v8f;

// ---------------------------------------------------------------------------
// Kernel 1: convert W [OUT_CH, IN_CH] f32 -> bf16 (row-major, K contiguous)
// ---------------------------------------------------------------------------
__global__ void convert_w_kernel(const float* __restrict__ W,
                                 bf16_t* __restrict__ Wb, int n) {
    int i = blockIdx.x * blockDim.x + threadIdx.x;
    if (i < n) Wb[i] = (bf16_t)W[i];
}

// ---------------------------------------------------------------------------
// Kernel 2: Xp = X @ W^T via v_wmma_f32_16x16x32_bf16.
// One wave per 16-row M-tile; 4 N-tiles of 16 cover OUT_CH=64; K-loop 8x32.
//
// CDNA5 16-bit A fragment (16x32, MxK), per lane:
//   M = lane&15, sel = lane>>4
//   elems 0..7  = K = k0 + sel*8 + 0..7
//   elems 8..15 = K = k0 + sel*8 + 16..23
// B fragment (32x16, KxN) is the mirror with N = lane&15 and B[k][n] = W[n][k],
// so both fragments are two contiguous 8-element runs along K in memory.
// C/D: VGPR v -> M = v + sel*8, N = lane&15.
// ---------------------------------------------------------------------------
__global__ __launch_bounds__(256) void gemm_xw_wmma_kernel(
    const float* __restrict__ X, const bf16_t* __restrict__ Wb,
    float* __restrict__ Xp, int nRows)
{
    const int lane  = threadIdx.x & 31;
    const int wave  = threadIdx.x >> 5;
    const int mTile = blockIdx.x * 8 + wave;            // 8 waves / block
    if (mTile * 16 >= nRows) return;                    // wave-uniform exit

    const int  mr   = lane & 15;
    const int  sel  = lane >> 4;                        // 0 or 1
    const float* xrow = X + (size_t)(mTile * 16 + mr) * IN_CH;

    v8f acc[4] = {};                                    // 4 N-tiles

    #pragma unroll
    for (int k0 = 0; k0 < IN_CH; k0 += 32) {
        // ---- A fragment: 16 f32 -> bf16, two contiguous 32B runs ----
        const float* pa = xrow + k0 + sel * 8;
        float4 a0 = *(const float4*)(pa + 0);
        float4 a1 = *(const float4*)(pa + 4);
        float4 a2 = *(const float4*)(pa + 16);
        float4 a3 = *(const float4*)(pa + 20);
        v16bf A;
        A[0]=(bf16_t)a0.x;  A[1]=(bf16_t)a0.y;  A[2]=(bf16_t)a0.z;  A[3]=(bf16_t)a0.w;
        A[4]=(bf16_t)a1.x;  A[5]=(bf16_t)a1.y;  A[6]=(bf16_t)a1.z;  A[7]=(bf16_t)a1.w;
        A[8]=(bf16_t)a2.x;  A[9]=(bf16_t)a2.y;  A[10]=(bf16_t)a2.z; A[11]=(bf16_t)a2.w;
        A[12]=(bf16_t)a3.x; A[13]=(bf16_t)a3.y; A[14]=(bf16_t)a3.z; A[15]=(bf16_t)a3.w;

        #pragma unroll
        for (int n = 0; n < 4; ++n) {
            const bf16_t* pb = Wb + (size_t)(n * 16 + mr) * IN_CH + k0 + sel * 8;
            v8bf b0 = *(const v8bf*)(pb + 0);           // K = kb..kb+7
            v8bf b1 = *(const v8bf*)(pb + 16);          // K = kb+16..kb+23
            v16bf B = __builtin_shufflevector(b0, b1,
                        0,1,2,3,4,5,6,7,8,9,10,11,12,13,14,15);
            // (neg_a, A, neg_b, B, c_mod, C, reuse_a, reuse_b)
            acc[n] = __builtin_amdgcn_wmma_f32_16x16x32_bf16(
                        false, A, false, B, (short)0, acc[n], false, false);
        }
    }

    // ---- store D: lane writes M = v + sel*8, N = nTile*16 + (lane&15) ----
    float* orow = Xp + (size_t)(mTile * 16) * OUT_CH;
    #pragma unroll
    for (int n = 0; n < 4; ++n) {
        #pragma unroll
        for (int v = 0; v < 8; ++v) {
            orow[(size_t)(v + sel * 8) * OUT_CH + n * 16 + mr] = acc[n][v];
        }
    }
}

// ---------------------------------------------------------------------------
// Kernel 3: zero output (harness poisons d_out with 0xAA)
// ---------------------------------------------------------------------------
__global__ void zero_kernel(float4* __restrict__ p, long long n4) {
    long long i = (long long)blockIdx.x * blockDim.x + threadIdx.x;
    if (i < n4) p[i] = make_float4(0.f, 0.f, 0.f, 0.f);
}

// ---------------------------------------------------------------------------
// Kernel 4: COO SpMM scatter-add. One wave per edge:
//   coalesced 256B gather of Xp[col], scale by val, 2 f32 atomics per lane
//   into out[row] (L2-resident -> global_atomic_add_f32, no CAS loop).
// ---------------------------------------------------------------------------
__global__ __launch_bounds__(256) void spmm_scatter_kernel(
    const long long* __restrict__ rows, const long long* __restrict__ cols,
    const float* __restrict__ vals, const float* __restrict__ Xp,
    float* __restrict__ out, long long nnz)
{
    const int lane = threadIdx.x & 31;
    const long long e = (long long)blockIdx.x * 8 + (threadIdx.x >> 5);
    if (e >= nnz) return;                               // wave-uniform exit

    const int   r = (int)rows[e];                       // broadcast loads
    const int   c = (int)cols[e];
    const float v = vals[e];

    const float2 g = *(const float2*)(Xp + (size_t)c * OUT_CH + lane * 2);
    float* dst = out + (size_t)r * OUT_CH + lane * 2;
    unsafeAtomicAdd(dst + 0, v * g.x);
    unsafeAtomicAdd(dst + 1, v * g.y);
}

// ---------------------------------------------------------------------------
// Host-side launcher
// inputs: [0]=g1, [1]=g2, [2]=X, [3]=W_lin, [4]=L_rows, [5]=L_cols, [6]=L_vals
// ---------------------------------------------------------------------------
extern "C" void kernel_launch(void* const* d_in, const int* in_sizes, int n_in,
                              void* d_out, int out_size, void* d_ws, size_t ws_size,
                              hipStream_t stream) {
    const float*     X     = (const float*)d_in[2];
    const float*     W     = (const float*)d_in[3];
    const long long* Lrows = (const long long*)d_in[4];
    const long long* Lcols = (const long long*)d_in[5];
    const float*     Lvals = (const float*)d_in[6];
    float*           out   = (float*)d_out;

    const int       nRows = in_sizes[2] / IN_CH;        // 100000
    const long long nnz   = in_sizes[4];                // 1600000
    const int       wElems = in_sizes[3];               // 64*256

    // workspace layout: [Wb bf16 : 32KB][Xp f32 : nRows*64*4]
    bf16_t* Wb = (bf16_t*)d_ws;
    float*  Xp = (float*)((char*)d_ws + 32768);

    // 1) W -> bf16
    convert_w_kernel<<<(wElems + 255) / 256, 256, 0, stream>>>(W, Wb, wElems);

    // 2) WMMA GEMM: 16-row tiles, 8 waves (tiles) per 256-thread block
    const int mTiles = (nRows + 15) / 16;
    gemm_xw_wmma_kernel<<<(mTiles + 7) / 8, 256, 0, stream>>>(X, Wb, Xp, nRows);

    // 3) zero output
    const long long n4 = (long long)nRows * OUT_CH / 4;
    zero_kernel<<<(unsigned)((n4 + 255) / 256), 256, 0, stream>>>((float4*)out, n4);

    // 4) edge scatter-add, one wave per edge (8 edges per block)
    spmm_scatter_kernel<<<(unsigned)((nnz + 7) / 8), 256, 0, stream>>>(
        Lrows, Lcols, Lvals, Xp, out, nnz);
}
